// TransformerBlock_41686952575368
// MI455X (gfx1250) — matrix operand order repaired and checked by reading, not compiled, not run on hardware
//
#include <hip/hip_runtime.h>
#include <hip/hip_bf16.h>

// ---------------------------------------------------------------------------
// Transformer block forward for gfx1250 (MI455X).
// bf16 WMMA (f32 accum), double-buffered LDS with async global->LDS copies,
// tile-parameterized GEMM: 128x128 block / 32x64 wave tile for big GEMMs.
// ---------------------------------------------------------------------------

#define TK 32
#define LDSS 40              // shorts per LDS row: 32 data + 8 pad (80B = 5*16B)

typedef __attribute__((ext_vector_type(16))) __bf16         v16bf;
typedef __attribute__((ext_vector_type(8)))  float          v8f;
typedef __attribute__((ext_vector_type(8)))  unsigned short us8;
typedef __attribute__((ext_vector_type(4)))  int            v4i;

union BFFrag { us8 u[2]; v16bf v; };

#if defined(__HIP_DEVICE_COMPILE__) && __has_builtin(__builtin_amdgcn_global_load_async_to_lds_b128)
#define ASYNC_LDS 1
typedef __attribute__((address_space(1))) v4i* g1_v4i_p;
typedef __attribute__((address_space(3))) v4i* l3_v4i_p;
#endif

__device__ __forceinline__ void async_copy16(const unsigned short* g, unsigned short* l) {
#ifdef ASYNC_LDS
  v4i* gg = (v4i*)const_cast<unsigned short*>(g);
  v4i* ll = (v4i*)l;
  __builtin_amdgcn_global_load_async_to_lds_b128((g1_v4i_p)gg, (l3_v4i_p)ll, 0, 0);
#else
  *(us8*)l = *(const us8*)g;
#endif
}

__device__ __forceinline__ void wait_async_lds() {
#ifdef ASYNC_LDS
 #if __has_builtin(__builtin_amdgcn_s_wait_asynccnt)
  __builtin_amdgcn_s_wait_asynccnt(0);
 #else
  asm volatile("s_wait_asynccnt 0" ::: "memory");
 #endif
#endif
}

__device__ __forceinline__ unsigned short f32_to_bf16_rne(float f) {
  unsigned int u = __float_as_uint(f);
  u += 0x7FFFu + ((u >> 16) & 1u);
  return (unsigned short)(u >> 16);
}

__device__ __forceinline__ float gelu_exact(float x) {
  return 0.5f * x * (1.0f + erff(x * 0.70710678118654752f));
}

// ---------------------------------------------------------------------------
// GEMM: C[M,N] = act( A[M,K](bf16) * B[N,K](bf16)^T + bias ) + addend
// OUTBF=1 -> C bf16 ; OUTBF=0 -> C fp32.
// Block tile BM x BN, K-step 32, wave grid WGM x WGN (wave tile BM/WGM x BN/WGN).
// ---------------------------------------------------------------------------
template<int OUTBF, int ACT, int BM, int BN, int WGM, int WGN>
__global__ __launch_bounds__(WGM * WGN * 32) void wmma_gemm_bf16(
    const unsigned short* __restrict__ A, int lda,
    const unsigned short* __restrict__ B, int ldb,
    void* __restrict__ Cv, int ldc,
    const float* __restrict__ bias,
    const float* __restrict__ addend,
    int K)
{
  constexpr int NT  = WGM * WGN * 32;     // threads per block
  constexpr int WTM = BM / WGM;           // wave tile M
  constexpr int WTN = BN / WGN;           // wave tile N
  constexpr int MI  = WTM / 16;           // A fragments per wave
  constexpr int NI  = WTN / 16;           // B fragments per wave
  constexpr int CHA = (BM * 4) / NT;      // 16B chunks per thread (A tile)
  constexpr int CHB = (BN * 4) / NT;      // 16B chunks per thread (B tile)

  __shared__ __attribute__((aligned(16))) unsigned short As[2][BM * LDSS];
  __shared__ __attribute__((aligned(16))) unsigned short Bs[2][BN * LDSS];

  const int tid  = threadIdx.x;
  const int lane = tid & 31;
  const int wv   = tid >> 5;
  const int bm   = blockIdx.y * BM;
  const int bn   = blockIdx.x * BN;
  const int wm   = (wv / WGN) * WTM;
  const int wn   = (wv % WGN) * WTN;

  v8f acc[MI][NI];
  #pragma unroll
  for (int i = 0; i < MI; ++i)
    #pragma unroll
    for (int j = 0; j < NI; ++j)
      #pragma unroll
      for (int r = 0; r < 8; ++r) acc[i][j][r] = 0.0f;

  const int m16 = lane & 15;
  const int kh  = lane >> 4;

  // stage one BMx32 (A) + BNx32 (B) bf16 tile pair into LDS buffer `buf`
  auto stageAB = [&](int buf, int k0) {
    #pragma unroll
    for (int i = 0; i < CHA; ++i) {
      int idx = tid + i * NT;
      int r   = idx >> 2;
      int c   = (idx & 3) * 8;
      async_copy16(&A[(long)(bm + r) * lda + k0 + c], &As[buf][r * LDSS + c]);
    }
    #pragma unroll
    for (int i = 0; i < CHB; ++i) {
      int idx = tid + i * NT;
      int r   = idx >> 2;
      int c   = (idx & 3) * 8;
      async_copy16(&B[(long)(bn + r) * ldb + k0 + c], &Bs[buf][r * LDSS + c]);
    }
  };

  const int nk = K / TK;
  stageAB(0, 0);

  for (int it = 0; it < nk; ++it) {
    wait_async_lds();
    __syncthreads();
    if (it + 1 < nk) stageAB((it + 1) & 1, (it + 1) * TK);
    const int cur = it & 1;

    // A 16x32 bf16: lanes 0-15 -> M=lane, K 0-7 & 16-23 ; lanes 16-31 -> K 8-15 & 24-31
    // B 32x16 bf16: lanes 0-15 -> N=lane, K 0-7 & 16-23 ; lanes 16-31 -> K 8-15 & 24-31
    BFFrag af[MI], bfr[NI];
    #pragma unroll
    for (int mi = 0; mi < MI; ++mi) {
      const unsigned short* p = &As[cur][(wm + mi * 16 + m16) * LDSS];
      af[mi].u[0] = *(const us8*)(p + kh * 8);
      af[mi].u[1] = *(const us8*)(p + 16 + kh * 8);
    }
    #pragma unroll
    for (int ni = 0; ni < NI; ++ni) {
      const unsigned short* p = &Bs[cur][(wn + ni * 16 + m16) * LDSS];
      bfr[ni].u[0] = *(const us8*)(p + kh * 8);
      bfr[ni].u[1] = *(const us8*)(p + 16 + kh * 8);
    }

    #pragma unroll
    for (int mi = 0; mi < MI; ++mi)
      #pragma unroll
      for (int ni = 0; ni < NI; ++ni)
        acc[mi][ni] = __builtin_amdgcn_wmma_f32_16x16x32_bf16(
            false, af[mi].v, false, bfr[ni].v, (short)0, acc[mi][ni],
            false, false);
  }

  // epilogue: C/D layout — VGPR r: M = r + 8*(lane>>4), N = lane&15
  float*          Cf = (float*)Cv;
  unsigned short* Cb = (unsigned short*)Cv;
  #pragma unroll
  for (int mi = 0; mi < MI; ++mi) {
    #pragma unroll
    for (int ni = 0; ni < NI; ++ni) {
      int gn = bn + wn + ni * 16 + m16;
      float bv = bias ? bias[gn] : 0.0f;
      #pragma unroll
      for (int r = 0; r < 8; ++r) {
        int  gm  = bm + wm + mi * 16 + kh * 8 + r;
        long off = (long)gm * ldc + gn;
        float v = acc[mi][ni][r] + bv;
        if (ACT) v = gelu_exact(v);
        if (addend) v += addend[off];
        if (OUTBF) Cb[off] = f32_to_bf16_rne(v);
        else       Cf[off] = v;
      }
    }
  }
}

// ---------------------------------------------------------------------------
// fp32 -> bf16 bulk conversion (8 elements / thread, b128 in, b128 out)
// ---------------------------------------------------------------------------
__global__ __launch_bounds__(256) void f32_to_bf16_k(
    const float* __restrict__ in, unsigned short* __restrict__ outp, long n)
{
  long i = ((long)blockIdx.x * 256 + threadIdx.x) * 8;
  if (i + 8 > n) return;
  float4 a = *(const float4*)(in + i);
  float4 b = *(const float4*)(in + i + 4);
  us8 o;
  o[0] = f32_to_bf16_rne(a.x); o[1] = f32_to_bf16_rne(a.y);
  o[2] = f32_to_bf16_rne(a.z); o[3] = f32_to_bf16_rne(a.w);
  o[4] = f32_to_bf16_rne(b.x); o[5] = f32_to_bf16_rne(b.y);
  o[6] = f32_to_bf16_rne(b.z); o[7] = f32_to_bf16_rne(b.w);
  *(us8*)(outp + i) = o;
}

// ---------------------------------------------------------------------------
// Batched bf16 transpose: out[b][c][r] = in[b][r][c]  (32x32 LDS tiles)
// ---------------------------------------------------------------------------
__global__ __launch_bounds__(256) void transpose_bf16_k(
    const unsigned short* __restrict__ in, unsigned short* __restrict__ outp,
    int rows, int cols)
{
  __shared__ unsigned short tile[32][33];
  const size_t base = (size_t)blockIdx.z * rows * cols;
  const int r0 = blockIdx.y * 32, c0 = blockIdx.x * 32;
  const int tx = threadIdx.x & 31, ty = threadIdx.x >> 5;   // 32 x 8
  #pragma unroll
  for (int i = 0; i < 32; i += 8)
    tile[ty + i][tx] = in[base + (size_t)(r0 + ty + i) * cols + c0 + tx];
  __syncthreads();
  #pragma unroll
  for (int i = 0; i < 32; i += 8)
    outp[base + (size_t)(c0 + ty + i) * rows + r0 + tx] = tile[tx][ty + i];
}

// ---------------------------------------------------------------------------
// Deterministic global sum / sum-of-squares (two stage, no atomics)
// ---------------------------------------------------------------------------
__global__ __launch_bounds__(256) void reduce_partial_k(
    const float* __restrict__ x, long n, float* __restrict__ part)
{
  __shared__ float r1[256], r2[256];
  float a = 0.f, b = 0.f;
  for (long i = (long)blockIdx.x * 256 + threadIdx.x; i < n;
       i += (long)gridDim.x * 256) {
    float v = x[i]; a += v; b += v * v;
  }
  r1[threadIdx.x] = a; r2[threadIdx.x] = b; __syncthreads();
  for (int o = 128; o > 0; o >>= 1) {
    if (threadIdx.x < o) {
      r1[threadIdx.x] += r1[threadIdx.x + o];
      r2[threadIdx.x] += r2[threadIdx.x + o];
    }
    __syncthreads();
  }
  if (threadIdx.x == 0) {
    part[blockIdx.x * 2 + 0] = r1[0];
    part[blockIdx.x * 2 + 1] = r2[0];
  }
}

__global__ __launch_bounds__(256) void reduce_final_k(
    const float* __restrict__ part, int nblocks, float* __restrict__ stats)
{
  __shared__ float r1[256], r2[256];
  float a = 0.f, b = 0.f;
  for (int i = threadIdx.x; i < nblocks; i += 256) {
    a += part[2 * i + 0]; b += part[2 * i + 1];
  }
  r1[threadIdx.x] = a; r2[threadIdx.x] = b; __syncthreads();
  for (int o = 128; o > 0; o >>= 1) {
    if (threadIdx.x < o) {
      r1[threadIdx.x] += r1[threadIdx.x + o];
      r2[threadIdx.x] += r2[threadIdx.x + o];
    }
    __syncthreads();
  }
  if (threadIdx.x == 0) { stats[0] = r1[0]; stats[1] = r2[0]; }
}

// ---------------------------------------------------------------------------
// Reference's LayerNorm: per-row mean, GLOBAL scalar variance (correction=0),
// x_norm = (x - mean)/var + eps, then scale/shift.  Output bf16 (feeds GEMM).
// ---------------------------------------------------------------------------
__global__ __launch_bounds__(256) void layernorm_gvar_k(
    const float* __restrict__ x, const float* __restrict__ scale,
    const float* __restrict__ shift, const float* __restrict__ stats,
    unsigned short* __restrict__ out, int cols, float invN)
{
  __shared__ float red[256];
  const long  row = blockIdx.x;
  const float* xr = x + row * cols;
  float s = 0.f;
  for (int c = threadIdx.x; c < cols; c += 256) s += xr[c];
  red[threadIdx.x] = s; __syncthreads();
  for (int o = 128; o > 0; o >>= 1) {
    if (threadIdx.x < o) red[threadIdx.x] += red[threadIdx.x + o];
    __syncthreads();
  }
  const float mean = red[0] / (float)cols;
  const float gm   = stats[0] * invN;
  const float var  = stats[1] * invN - gm * gm;
  const float inv  = 1.0f / var;
  unsigned short* orow = out + row * cols;
  for (int c = threadIdx.x; c < cols; c += 256)
    orow[c] = f32_to_bf16_rne(((xr[c] - mean) * inv + 1e-5f) * scale[c] + shift[c]);
}

// ---------------------------------------------------------------------------
// Causal softmax: fp32 scores in, bf16 probabilities out, row = blockIdx.x.
// ---------------------------------------------------------------------------
__global__ __launch_bounds__(256) void softmax_causal_k(
    const float* __restrict__ sc, unsigned short* __restrict__ pr,
    int T, float scale)
{
  __shared__ float red[256];
  const int t   = blockIdx.x;
  const float* row = sc + (long)t * T;
  unsigned short* prow = pr + (long)t * T;
  const int lim = t + 1;
  const int tid = threadIdx.x;

  float m = -3.0e38f;
  for (int s = tid; s < lim; s += 256) m = fmaxf(m, row[s] * scale);
  red[tid] = m; __syncthreads();
  for (int o = 128; o > 0; o >>= 1) {
    if (tid < o) red[tid] = fmaxf(red[tid], red[tid + o]);
    __syncthreads();
  }
  m = red[0]; __syncthreads();

  float sum = 0.f;
  for (int s = tid; s < lim; s += 256) sum += __expf(row[s] * scale - m);
  red[tid] = sum; __syncthreads();
  for (int o = 128; o > 0; o >>= 1) {
    if (tid < o) red[tid] += red[tid + o];
    __syncthreads();
  }
  const float inv = 1.0f / red[0];

  for (int s = tid; s < T; s += 256)
    prow[s] = (s < lim) ? f32_to_bf16_rne(__expf(row[s] * scale - m) * inv)
                        : (unsigned short)0;
}

// ---------------------------------------------------------------------------
// Host-side orchestration
// ---------------------------------------------------------------------------
extern "C" void kernel_launch(void* const* d_in, const int* in_sizes, int n_in,
                              void* d_out, int out_size, void* d_ws, size_t ws_size,
                              hipStream_t stream) {
  (void)in_sizes; (void)n_in; (void)out_size; (void)ws_size;

  const int Bb = 4, T = 2048, EMB = 1024, H = 16, D = 64, DFF = 4096;
  const int M  = Bb * T;                 // 8192 rows
  const long NE = (long)M * EMB;

  const float* x   = (const float*)d_in[0];
  const float* Wq  = (const float*)d_in[1];
  const float* bq  = (const float*)d_in[2];
  const float* Wk  = (const float*)d_in[3];
  const float* bk  = (const float*)d_in[4];
  const float* Wv  = (const float*)d_in[5];
  const float* bv  = (const float*)d_in[6];
  const float* Wo  = (const float*)d_in[7];
  const float* bo  = (const float*)d_in[8];
  const float* l1w = (const float*)d_in[9];
  const float* l1b = (const float*)d_in[10];
  const float* l2w = (const float*)d_in[11];
  const float* l2b = (const float*)d_in[12];
  const float* n1s = (const float*)d_in[13];
  const float* n1b = (const float*)d_in[14];
  const float* n2s = (const float*)d_in[15];
  const float* n2b = (const float*)d_in[16];
  float* out = (float*)d_out;

  // workspace layout
  char* wsp = (char*)d_ws;
  auto alloc = [&](size_t bytes) {
    char* p = wsp; wsp += (bytes + 255) & ~(size_t)255; return p;
  };
  float*          stats = (float*)alloc(256);
  float*          part  = (float*)alloc(4096);
  unsigned short* wqb   = (unsigned short*)alloc((size_t)EMB * EMB * 2);
  unsigned short* wkb   = (unsigned short*)alloc((size_t)EMB * EMB * 2);
  unsigned short* wvb   = (unsigned short*)alloc((size_t)EMB * EMB * 2);
  unsigned short* wob   = (unsigned short*)alloc((size_t)EMB * EMB * 2);
  unsigned short* l1b16 = (unsigned short*)alloc((size_t)DFF * EMB * 2);
  unsigned short* l2b16 = (unsigned short*)alloc((size_t)EMB * DFF * 2);
  unsigned short* hbf   = (unsigned short*)alloc((size_t)NE * 2);   // ln1 out
  unsigned short* qbf   = (unsigned short*)alloc((size_t)NE * 2);
  unsigned short* kbf   = (unsigned short*)alloc((size_t)NE * 2);
  unsigned short* vbf   = (unsigned short*)alloc((size_t)NE * 2);
  unsigned short* vT    = (unsigned short*)alloc((size_t)NE * 2);   // [b][c][t]
  unsigned short* ctxb  = (unsigned short*)alloc((size_t)NE * 2);
  float*          x2    = (float*)alloc((size_t)NE * 4);
  unsigned short* h2bf  = (unsigned short*)alloc((size_t)NE * 2);
  unsigned short* ff1b  = (unsigned short*)alloc((size_t)M * DFF * 2);
  float*          scb   = (float*)alloc((size_t)T * T * 4);
  unsigned short* pb    = (unsigned short*)alloc((size_t)T * T * 2);

  const dim3 blkB(256);   // 128x128 tile config (8 waves)
  const dim3 blkS(128);   // 64x64 tile config (4 waves)
  const float invN = 1.0f / (float)NE;

  // ---- one-time weight conversion fp32 -> bf16 ----------------------------
  {
    const long nw = (long)EMB * EMB;
    f32_to_bf16_k<<<(unsigned)(nw / 2048), 256, 0, stream>>>(Wq, wqb, nw);
    f32_to_bf16_k<<<(unsigned)(nw / 2048), 256, 0, stream>>>(Wk, wkb, nw);
    f32_to_bf16_k<<<(unsigned)(nw / 2048), 256, 0, stream>>>(Wv, wvb, nw);
    f32_to_bf16_k<<<(unsigned)(nw / 2048), 256, 0, stream>>>(Wo, wob, nw);
    const long nf = (long)DFF * EMB;
    f32_to_bf16_k<<<(unsigned)(nf / 2048), 256, 0, stream>>>(l1w, l1b16, nf);
    f32_to_bf16_k<<<(unsigned)(nf / 2048), 256, 0, stream>>>(l2w, l2b16, nf);
  }

  // ---- LayerNorm 1 --------------------------------------------------------
  reduce_partial_k<<<512, 256, 0, stream>>>(x, NE, part);
  reduce_final_k<<<1, 256, 0, stream>>>(part, 512, stats);
  layernorm_gvar_k<<<M, 256, 0, stream>>>(x, n1s, n1b, stats, hbf, EMB, invN);

  // ---- Q, K, V projections (bf16 out) -------------------------------------
  const dim3 gproj(EMB / 128, M / 128);
  wmma_gemm_bf16<1, 0, 128, 128, 4, 2><<<gproj, blkB, 0, stream>>>(
      hbf, EMB, wqb, EMB, qbf, EMB, bq, nullptr, EMB);
  wmma_gemm_bf16<1, 0, 128, 128, 4, 2><<<gproj, blkB, 0, stream>>>(
      hbf, EMB, wkb, EMB, kbf, EMB, bk, nullptr, EMB);
  wmma_gemm_bf16<1, 0, 128, 128, 4, 2><<<gproj, blkB, 0, stream>>>(
      hbf, EMB, wvb, EMB, vbf, EMB, bv, nullptr, EMB);

  // ---- V transpose: vT[b][c][t] = v[b][t][c] ------------------------------
  transpose_bf16_k<<<dim3(EMB / 32, T / 32, Bb), 256, 0, stream>>>(vbf, vT, T, EMB);

  // ---- Attention, one (batch, head) score chunk at a time -----------------
  for (int b = 0; b < Bb; ++b) {
    for (int hh = 0; hh < H; ++hh) {
      const unsigned short* qh = qbf + (size_t)b * T * EMB + hh * D;
      const unsigned short* kh = kbf + (size_t)b * T * EMB + hh * D;
      const unsigned short* vh = vT + ((size_t)b * EMB + hh * D) * T;  // [64][T]
      unsigned short*       ch = ctxb + (size_t)b * T * EMB + hh * D;

      // scores = q @ k^T  (M=T, N=T, K=64), fp32 out
      wmma_gemm_bf16<0, 0, 128, 128, 4, 2><<<dim3(T / 128, T / 128), blkB, 0, stream>>>(
          qh, EMB, kh, EMB, scb, T, nullptr, nullptr, D);
      // causal softmax (scale 1/sqrt(64)), bf16 probabilities out
      softmax_causal_k<<<T, 256, 0, stream>>>(scb, pb, T, 0.125f);
      // ctx = P @ vT^T  (M=T, N=64, K=T), bf16 out, narrow tile config
      wmma_gemm_bf16<1, 0, 64, 64, 2, 2><<<dim3(D / 64, T / 64), blkS, 0, stream>>>(
          pb, T, vh, T, ch, EMB, nullptr, nullptr, T);
    }
  }

  // ---- Output projection + residual (fp32 out) ----------------------------
  wmma_gemm_bf16<0, 0, 128, 128, 4, 2><<<gproj, blkB, 0, stream>>>(
      ctxb, EMB, wob, EMB, x2, EMB, bo, x, EMB);

  // ---- LayerNorm 2 --------------------------------------------------------
  reduce_partial_k<<<512, 256, 0, stream>>>(x2, NE, part);
  reduce_final_k<<<1, 256, 0, stream>>>(part, 512, stats + 2);
  layernorm_gvar_k<<<M, 256, 0, stream>>>(x2, n2s, n2b, stats + 2, h2bf, EMB, invN);

  // ---- FFN ----------------------------------------------------------------
  wmma_gemm_bf16<1, 1, 128, 128, 4, 2><<<dim3(DFF / 128, M / 128), blkB, 0, stream>>>(
      h2bf, EMB, l1b16, EMB, ff1b, DFF, l1b, nullptr, EMB);
  wmma_gemm_bf16<0, 0, 128, 128, 4, 2><<<dim3(EMB / 128, M / 128), blkB, 0, stream>>>(
      ff1b, DFF, l2b16, DFF, out, EMB, l2b, x2, DFF);
}
